// LeNet5Binary_72507637891896
// MI455X (gfx1250) — matrix operand order repaired and checked by reading, not compiled, not run on hardware
//
#include <hip/hip_runtime.h>
#include <hip/hip_bf16.h>

typedef __attribute__((ext_vector_type(16))) _Float16 v16h;
typedef __attribute__((ext_vector_type(8)))  float    v8f;
typedef __attribute__((ext_vector_type(4)))  int      v4i;

// Fragment order (CDNA5 ISA 7.12.2, wave32, 16x16x32 f16):
//   A 16x32: lane holds row M=lane%16; elem idx -> K = (lane>>4)*16 + idx as stored here
//   B 32x16: lane holds col N=lane%16; elem idx -> K = (lane>>4)*16 + idx as stored here (same as A)
//   C/D    : lane -> N=lane%16; elem i -> M = i + 8*(lane>>4)
// Swizzled operand storage: [tile][kstep][lane][16 halfs] so each lane's
// fragment is one contiguous, 32B-aligned v16h.

__device__ __forceinline__ float sgnf(float v) {
  return v > 0.f ? 1.f : (v < 0.f ? -1.f : 0.f);
}
__device__ __forceinline__ v16h ldv16(const _Float16* p) {
  return *(const v16h*)p;
}

// ---- async global->LDS 16B chunk copy (CDNA5 GLOBAL_LOAD_ASYNC_TO_LDS_B128) -
#if __has_builtin(__builtin_amdgcn_global_load_async_to_lds_b128) && \
    __has_builtin(__builtin_amdgcn_s_wait_asynccnt)
#define USE_ASYNC_LDS 1
#else
#define USE_ASYNC_LDS 0
#endif

__device__ __forceinline__ void copy16_g2l(void* lds, const void* g) {
#if USE_ASYNC_LDS
  __builtin_amdgcn_global_load_async_to_lds_b128(
      (__attribute__((address_space(1))) v4i*)(void*)g,
      (__attribute__((address_space(3))) v4i*)lds, 0, 0);
#else
  *(float4*)lds = *(const float4*)g;
#endif
}
__device__ __forceinline__ void async_join() {
#if USE_ASYNC_LDS
  __builtin_amdgcn_s_wait_asynccnt(0);
#endif
}

// stats buffer layout (1024 floats): [0..255]=sum, [256..511]=sumsq,
// [512..767]=scale, [768..1023]=bias.  channel bases: bn1=0, bn2=8, bn3=32.

__global__ void k_zero(float* stats) {
  int i = blockIdx.x * 256 + threadIdx.x;
  if (i < 1024) stats[i] = 0.f;
}

// Binarize + swizzle weights into fragment-ready f16 with zero padding.
__global__ void k_prep(const float* __restrict__ w2, const float* __restrict__ w3,
                       const float* __restrict__ fw1, const float* __restrict__ fw2,
                       _Float16* w2s, _Float16* w3s, _Float16* wf1s, _Float16* wf2p) {
  int i = blockIdx.x * blockDim.x + threadIdx.x;
  int stride = gridDim.x * blockDim.x;
  // conv2: N=16 (1 tile), K=160(150): [s<5][lane<32][16]
  for (int t = i; t < 2560; t += stride) {
    int idx = t & 15, lane = (t >> 4) & 31, s = t >> 9;
    int k = s * 32 + ((lane >> 4) << 4) + idx;
    int n = lane & 15;
    w2s[t] = (_Float16)((k < 150) ? sgnf(w2[n * 150 + k]) : 0.f);
  }
  // conv3: N=128(120), K=416(400): [nt<8][s<13][lane][16]
  for (int t = i; t < 53248; t += stride) {
    int idx = t & 15, lane = (t >> 4) & 31, g = t >> 9;
    int s = g % 13, nt = g / 13;
    int k = s * 32 + ((lane >> 4) << 4) + idx;
    int n = nt * 16 + (lane & 15);
    w3s[t] = (_Float16)((k < 400 && n < 120) ? sgnf(w3[n * 400 + k]) : 0.f);
  }
  // fc1: N=96(84), K=128(120): [nt<6][s<4][lane][16]
  for (int t = i; t < 12288; t += stride) {
    int idx = t & 15, lane = (t >> 4) & 31, g = t >> 9;
    int s = g & 3, nt = g >> 2;
    int k = s * 32 + ((lane >> 4) << 4) + idx;
    int n = nt * 16 + (lane & 15);
    wf1s[t] = (_Float16)((k < 120 && n < 84) ? sgnf(fw1[n * 120 + k]) : 0.f);
  }
  for (int t = i; t < 840; t += stride)  // fc2 kept full precision
    wf2p[t] = (_Float16)fw2[t];
}

// ---- layer 1: conv 5x5 (1 ch), batch stats pass -----------------------------
__global__ void k_conv1_stats(const float* __restrict__ x, const float* __restrict__ w1,
                              const float* __restrict__ b1, float* stats) {
  __shared__ __align__(16) float sx[1024];
  __shared__ float sw[152];
  __shared__ float ssum[8], ssq[8];
  const int img = blockIdx.x, t = threadIdx.x;
  copy16_g2l(sx + t * 4, x + (size_t)img * 1024 + t * 4);  // 256 chunks
  if (t < 150) sw[t] = w1[t];
  if (t < 8) { ssum[t] = 0.f; ssq[t] = 0.f; }
  async_join();
  __syncthreads();
  for (int c = 0; c < 6; ++c) {
    float bias = b1[c];
    float ls = 0.f, lq = 0.f;
    for (int p = t; p < 784; p += 256) {
      int oy = p / 28, ox = p % 28;
      float acc = bias;
#pragma unroll
      for (int ky = 0; ky < 5; ++ky)
#pragma unroll
        for (int kx = 0; kx < 5; ++kx)
          acc += sx[(oy + ky) * 32 + ox + kx] * sw[c * 25 + ky * 5 + kx];
      ls += acc; lq += acc * acc;
    }
    atomicAdd(&ssum[c], ls); atomicAdd(&ssq[c], lq);
  }
  __syncthreads();
  if (t < 6) { atomicAdd(&stats[t], ssum[t]); atomicAdd(&stats[256 + t], ssq[t]); }
}

// finalize BN: scale = g*rsqrt(var+eps), bias = beta - mean*scale
__global__ void k_bn_final(float* stats, int base, int nch, float inv_count, float eps,
                           const float* gamma, const float* beta) {
  int i = threadIdx.x;
  if (i >= nch) return;
  float mean = stats[base + i] * inv_count;
  float var  = stats[256 + base + i] * inv_count - mean * mean;
  float inv  = rsqrtf(var + eps);
  float g = 1.f, b = 0.f;
  if (gamma) { g = fmaxf(gamma[i], 0.01f); b = beta[i]; }
  float sc = g * inv;
  stats[512 + base + i] = sc;
  stats[768 + base + i] = b - mean * sc;
}

// ---- layer 1 apply: recompute conv, pool(max) then BN+relu (scale>0 commutes)
__global__ void k_conv1_apply(const float* __restrict__ x, const float* __restrict__ w1,
                              const float* __restrict__ b1, const float* __restrict__ stats,
                              _Float16* h1) {
  __shared__ __align__(16) float sx[1024];
  __shared__ float sw[152];
  __shared__ float sy[4704];
  const int img = blockIdx.x, t = threadIdx.x;
  copy16_g2l(sx + t * 4, x + (size_t)img * 1024 + t * 4);
  if (t < 150) sw[t] = w1[t];
  async_join();
  __syncthreads();
  for (int i = t; i < 4704; i += 256) {
    int c = i / 784, p = i % 784, oy = p / 28, ox = p % 28;
    float acc = b1[c];
#pragma unroll
    for (int ky = 0; ky < 5; ++ky)
#pragma unroll
      for (int kx = 0; kx < 5; ++kx)
        acc += sx[(oy + ky) * 32 + ox + kx] * sw[c * 25 + ky * 5 + kx];
    sy[i] = acc;
  }
  __syncthreads();
  for (int i = t; i < 1176; i += 256) {
    int c = i / 196, r = i % 196, py = r / 14, px = r % 14;
    int b0 = c * 784 + (2 * py) * 28 + 2 * px;
    float m = fmaxf(fmaxf(sy[b0], sy[b0 + 1]), fmaxf(sy[b0 + 28], sy[b0 + 29]));
    float v = fmaxf(m * stats[512 + c] + stats[768 + c], 0.f);
    h1[(size_t)img * 1176 + i] = (_Float16)v;
  }
}

// ---- layer 2: binary conv as implicit GEMM.  M=112(100) px, N=16, K=160(150)
template <bool STATS>
__global__ void k_conv2(const _Float16* __restrict__ h1g, const _Float16* __restrict__ w2s,
                        const float* __restrict__ b2, float* stats, _Float16* h2) {
  __shared__ __align__(16) _Float16 sh1[1176];
  __shared__ __align__(32) _Float16 sAs[17920];   // [mt<7][s<5][lane][16]
  __shared__ __align__(32) _Float16 sw2[2560];    // [s<5][lane][16]
  __shared__ float sy2[1600];
  __shared__ float ssum[16], ssq[16];
  const int img = blockIdx.x, t = threadIdx.x;
  for (int c = t; c < 147; c += 256) copy16_g2l(sh1 + c * 8, h1g + (size_t)img * 1176 + c * 8);
  for (int c = t; c < 320; c += 256) copy16_g2l(sw2 + c * 8, w2s + c * 8);
  if (STATS && t < 16) { ssum[t] = 0.f; ssq[t] = 0.f; }
  async_join();
  __syncthreads();
  // build im2col A directly in fragment order
  for (int i = t; i < 17920; i += 256) {
    int idx = i & 15, lane = (i >> 4) & 31, g = i >> 9;   // g = mt*5 + s
    int s = g % 5, mt = g / 5;
    int m = mt * 16 + (lane & 15);
    int k = s * 32 + ((lane >> 4) << 4) + idx;
    float v = 0.f;
    if (m < 100 && k < 150) {
      int c2 = k / 25, r = k % 25;
      int oy = m / 10, ox = m % 10;
      v = (float)sh1[c2 * 196 + (oy + r / 5) * 14 + (ox + r % 5)];
    }
    sAs[i] = (_Float16)v;
  }
  __syncthreads();
  const int wave = t >> 5, lane = t & 31, half_id = lane >> 4, l16 = lane & 15;
  if (wave < 7) {
    v8f acc = {};
    for (int s = 0; s < 5; ++s) {
      v16h a = ldv16(&sAs[((wave * 5 + s) * 32 + lane) << 4]);
      v16h b = ldv16(&sw2[((s * 32 + lane)) << 4]);
      acc = __builtin_amdgcn_wmma_f32_16x16x32_f16(false, a, false, b, (short)0, acc,
                                                   false, false);
    }
    const int n = l16;
    const float bias = b2[n];
    if (STATS) {
      float ls = 0.f, lq = 0.f;
#pragma unroll
      for (int i = 0; i < 8; ++i) {
        int mm = wave * 16 + i + 8 * half_id;
        if (mm < 100) { float v = acc[i] + bias; ls += v; lq += v * v; }
      }
      atomicAdd(&ssum[n], ls); atomicAdd(&ssq[n], lq);
    } else {
      float sc = stats[512 + 8 + n], bo = stats[768 + 8 + n];
#pragma unroll
      for (int i = 0; i < 8; ++i) {
        int mm = wave * 16 + i + 8 * half_id;
        if (mm < 100) sy2[n * 100 + mm] = fmaxf((acc[i] + bias) * sc + bo, 0.f);
      }
    }
  }
  __syncthreads();
  if (STATS) {
    if (t < 16) { atomicAdd(&stats[8 + t], ssum[t]); atomicAdd(&stats[256 + 8 + t], ssq[t]); }
  } else {
    for (int i = t; i < 416; i += 256) {       // pool 10x10 -> 5x5, K-padded layout
      float v = 0.f;
      if (i < 400) {
        int c = i / 25, r = i % 25, py = r / 5, px = r % 5;
        int b0 = c * 100 + (2 * py) * 10 + 2 * px;
        v = fmaxf(fmaxf(sy2[b0], sy2[b0 + 1]), fmaxf(sy2[b0 + 10], sy2[b0 + 11]));
      }
      h2[(size_t)img * 416 + i] = (_Float16)v;
    }
  }
}

// ---- layer 3: pure GEMM  M=8192 imgs, N=128(120), K=416(400) ----------------
__global__ void k_conv3(const _Float16* __restrict__ h2, const _Float16* __restrict__ w3s,
                        const float* __restrict__ b3, float* stats, float* y3) {
  __shared__ __align__(32) _Float16 sAs[6656];   // [s<13][lane][16], shared by 8 waves
  __shared__ float ssum[128], ssq[128];
  const int mt = blockIdx.x, t = threadIdx.x;
  const size_t base = (size_t)mt * 16 * 416;
  // stage A in fragment order: pure 16B chunk permutation (k runs of 8 stay contiguous)
  for (int c = t; c < 832; c += 256) {
    int idx8 = c & 1, lane = (c >> 1) & 31, s = c >> 6;
    int m = lane & 15, h = lane >> 4;
    int kst = s * 32 + h * 16 + idx8 * 8;
    copy16_g2l(sAs + c * 8, h2 + base + m * 416 + kst);
  }
  if (t < 128) { ssum[t] = 0.f; ssq[t] = 0.f; }
  async_join();
  __syncthreads();
  const int wave = t >> 5, lane = t & 31, half_id = lane >> 4, l16 = lane & 15;
  const int ntile = wave;
  v8f acc = {};
  for (int s = 0; s < 13; ++s) {
    v16h a = ldv16(&sAs[(s * 32 + lane) << 4]);
    v16h b = *(const v16h*)(w3s + ((size_t)((ntile * 13 + s) * 32 + lane) << 4));
    acc = __builtin_amdgcn_wmma_f32_16x16x32_f16(false, a, false, b, (short)0, acc,
                                                 false, false);
  }
  const int n = ntile * 16 + l16;
  if (n < 120) {
    const float bias = b3[n];
    float ls = 0.f, lq = 0.f;
#pragma unroll
    for (int i = 0; i < 8; ++i) {
      int img = mt * 16 + i + 8 * half_id;
      float v = acc[i] + bias;
      y3[(size_t)img * 120 + n] = v;
      ls += v; lq += v * v;
    }
    atomicAdd(&ssum[n], ls); atomicAdd(&ssq[n], lq);
  }
  __syncthreads();
  if (t < 120) { atomicAdd(&stats[32 + t], ssum[t]); atomicAdd(&stats[256 + 32 + t], ssq[t]); }
}

// BN3 + relu -> f16, K padded 120 -> 128 for fc1
__global__ void k_apply3(const float* __restrict__ y3, const float* __restrict__ stats,
                         _Float16* h3) {
  int i = blockIdx.x * 256 + threadIdx.x;      // 8192*128 total
  int img = i >> 7, k = i & 127;
  float v = 0.f;
  if (k < 120)
    v = fmaxf(y3[(size_t)img * 120 + k] * stats[512 + 32 + k] + stats[768 + 32 + k], 0.f);
  h3[i] = (_Float16)v;
}

// ---- layer 4: fc1 via WMMA (M=16/block, N=96(84), K=128(120)); fc2 scalar ---
__global__ void k_fc(const _Float16* __restrict__ h3, const _Float16* __restrict__ wf1s,
                     const float* __restrict__ fb1, const _Float16* __restrict__ wf2p,
                     const float* __restrict__ fb2, float* __restrict__ out) {
  __shared__ __align__(32) _Float16 sAs[2048];    // [s<4][lane][16]
  __shared__ __align__(32) _Float16 sWs[12288];   // [nt<6][s<4][lane][16]
  __shared__ float sH[16 * 96];
  const int mt = blockIdx.x, t = threadIdx.x;
  for (int c = t; c < 256; c += 256) {            // A tile, fragment order
    int idx8 = c & 1, lane = (c >> 1) & 31, s = c >> 6;
    int m = lane & 15, h = lane >> 4;
    int kst = s * 32 + h * 16 + idx8 * 8;
    copy16_g2l(sAs + c * 8, h3 + (size_t)mt * 2048 + m * 128 + kst);
  }
  for (int c = t; c < 1536; c += 256) copy16_g2l(sWs + c * 8, wf1s + c * 8);
  async_join();
  __syncthreads();
  const int wave = t >> 5, lane = t & 31, half_id = lane >> 4, l16 = lane & 15;
  if (wave < 6) {
    v8f acc = {};
    for (int s = 0; s < 4; ++s) {
      v16h a = ldv16(&sAs[(s * 32 + lane) << 4]);
      v16h b = ldv16(&sWs[((wave * 4 + s) * 32 + lane) << 4]);
      acc = __builtin_amdgcn_wmma_f32_16x16x32_f16(false, a, false, b, (short)0, acc,
                                                   false, false);
    }
    const int n = wave * 16 + l16;
    const float bias = (n < 84) ? fb1[n] : 0.f;
#pragma unroll
    for (int i = 0; i < 8; ++i) {
      int m = i + 8 * half_id;
      sH[m * 96 + n] = (n < 84) ? fmaxf(acc[i] + bias, 0.f) : 0.f;
    }
  }
  __syncthreads();
  if (t < 160) {                                  // fc2: 16 imgs x 10 outs
    int m = t / 10, j = t % 10;
    float s = fb2[j];
    for (int k = 0; k < 84; ++k) s += sH[m * 96 + k] * (float)wf2p[j * 84 + k];
    out[((size_t)mt * 16 + m) * 10 + j] = s;
  }
}

extern "C" void kernel_launch(void* const* d_in, const int* in_sizes, int n_in,
                              void* d_out, int out_size, void* d_ws, size_t ws_size,
                              hipStream_t stream) {
  (void)in_sizes; (void)n_in; (void)out_size; (void)ws_size;
  const float* x   = (const float*)d_in[0];
  const float* w1  = (const float*)d_in[1];
  const float* b1  = (const float*)d_in[2];
  const float* w2  = (const float*)d_in[3];
  const float* b2  = (const float*)d_in[4];
  const float* g2  = (const float*)d_in[5];
  const float* be2 = (const float*)d_in[6];
  const float* w3  = (const float*)d_in[7];
  const float* b3  = (const float*)d_in[8];
  const float* g3  = (const float*)d_in[9];
  const float* be3 = (const float*)d_in[10];
  const float* fw1 = (const float*)d_in[11];
  const float* fb1 = (const float*)d_in[12];
  const float* fw2 = (const float*)d_in[13];
  const float* fb2 = (const float*)d_in[14];
  float* out = (float*)d_out;

  char* ws = (char*)d_ws;
  size_t off = 0;
  auto take = [&](size_t bytes) -> char* {
    char* p = ws + off;
    off = (off + bytes + 255) & ~(size_t)255;
    return p;
  };
  float*    stats = (float*)   take(1024 * 4);
  _Float16* w2s   = (_Float16*)take(2560 * 2);
  _Float16* w3s   = (_Float16*)take(53248 * 2);
  _Float16* wf1s  = (_Float16*)take(12288 * 2);
  _Float16* wf2p  = (_Float16*)take(840 * 2);
  _Float16* h1    = (_Float16*)take((size_t)8192 * 1176 * 2);
  _Float16* h2    = (_Float16*)take((size_t)8192 * 416 * 2);
  float*    y3    = (float*)   take((size_t)8192 * 120 * 4);
  _Float16* h3    = (_Float16*)take((size_t)8192 * 128 * 2);

  k_zero<<<4, 256, 0, stream>>>(stats);
  k_prep<<<64, 256, 0, stream>>>(w2, w3, fw1, fw2, w2s, w3s, wf1s, wf2p);

  k_conv1_stats<<<8192, 256, 0, stream>>>(x, w1, b1, stats);
  k_bn_final<<<1, 128, 0, stream>>>(stats, 0, 6, 1.f / (8192.f * 784.f), 1e-4f,
                                    nullptr, nullptr);
  k_conv1_apply<<<8192, 256, 0, stream>>>(x, w1, b1, stats, h1);

  k_conv2<true><<<8192, 256, 0, stream>>>(h1, w2s, b2, stats, h2);
  k_bn_final<<<1, 128, 0, stream>>>(stats, 8, 16, 1.f / (8192.f * 100.f), 1e-4f, g2, be2);
  k_conv2<false><<<8192, 256, 0, stream>>>(h1, w2s, b2, stats, h2);

  k_conv3<<<512, 256, 0, stream>>>(h2, w3s, b3, stats, y3);
  k_bn_final<<<1, 128, 0, stream>>>(stats, 32, 120, 1.f / 8192.f, 1e-5f, g3, be3);
  k_apply3<<<4096, 256, 0, stream>>>(y3, stats, h3);

  k_fc<<<512, 256, 0, stream>>>(h3, wf1s, fb1, wf2p, fb2, out);
}
